// MultiHeadAttention_62766652064237
// MI455X (gfx1250) — matrix operand order repaired and checked by reading, not compiled, not run on hardware
//
#include <hip/hip_runtime.h>
#include <hip/hip_bf16.h>

// MultiHeadAttention for MI455X (gfx1250): bf16 WMMA everywhere,
// async-to-LDS tile staging where the toolchain exposes it.
// B=4, S=2048, E=1024, H=16, D=64.

#define BB 4
#define SS 2048
#define EE 1024
#define HH 16
#define DD 64

typedef __bf16 bf16;
typedef __attribute__((ext_vector_type(16))) __bf16 v16bf;
typedef __attribute__((ext_vector_type(8)))  __bf16 bf16x8;
typedef __attribute__((ext_vector_type(4)))  __bf16 bf16x4;
typedef __attribute__((ext_vector_type(8)))  float  v8f;
typedef __attribute__((ext_vector_type(4)))  float  f32x4;
typedef __attribute__((ext_vector_type(4)))  int    v4i;

#define NEG_INF (-__builtin_inff())

// ---------------------------------------------------------------------------
// Async global->LDS copy (CDNA5 GLOBAL_LOAD_ASYNC_TO_LDS_B128, ASYNCcnt).
// Guarded: falls back to VGPR staging if the builtin is unavailable.
// Builtin signature (per hipcc diagnostic): pointers are v4i* (generic).
// ---------------------------------------------------------------------------
#if defined(__has_builtin)
#if __has_builtin(__builtin_amdgcn_global_load_async_to_lds_b128)
#define HAVE_ASYNC_LDS 1
#endif
#endif

#ifdef HAVE_ASYNC_LDS
static __device__ __forceinline__ void async_copy_b128(void* lds_dst,
                                                       const void* g_src) {
  __builtin_amdgcn_global_load_async_to_lds_b128((v4i*)g_src, (v4i*)lds_dst,
                                                 0, 0);
}
static __device__ __forceinline__ void wait_async0() {
#if __has_builtin(__builtin_amdgcn_s_wait_asynccnt)
  __builtin_amdgcn_s_wait_asynccnt(0);
#else
  asm volatile("s_wait_asynccnt 0x0" ::: "memory");
#endif
}
static __device__ __forceinline__ void wait_async1() {
#if __has_builtin(__builtin_amdgcn_s_wait_asynccnt)
  __builtin_amdgcn_s_wait_asynccnt(1);
#else
  asm volatile("s_wait_asynccnt 0x1" ::: "memory");
#endif
}
#endif

// ---------------------------------------------------------------------------
// WMMA wrapper: D = A(16x32 bf16) * B(32x16 bf16) + C(16x16 f32)
// ---------------------------------------------------------------------------
static __device__ __forceinline__ v8f wmma_bf16(v16bf a, v16bf b, v8f c) {
  return __builtin_amdgcn_wmma_f32_16x16x32_bf16(
      /*neg_a=*/false, a, /*neg_b=*/false, b,
      /*c_mod=*/(short)0, c, /*reuse_a=*/false, /*reuse_b=*/false);
}

// A-fragment (16x32, 16-bit): lane L holds row m=L%16; element j maps to
// k = kbase + (j<8 ? (L/16)*8 + j : 16 + (L/16)*8 + (j-8)).
static __device__ __forceinline__ v16bf afrag_bf16(const bf16* row, int kbase, int lh) {
  bf16x8 lo = *(const bf16x8*)(row + kbase + lh * 8);
  bf16x8 hi = *(const bf16x8*)(row + kbase + 16 + lh * 8);
  v16bf r;
#pragma unroll
  for (int i = 0; i < 8; ++i) { r[i] = lo[i]; r[8 + i] = hi[i]; }
  return r;
}

// Same A-fragment sourced from fp32 memory (convert to bf16 on load).
static __device__ __forceinline__ v16bf afrag_f32(const float* row, int kbase, int lh) {
  const f32x4* p0 = (const f32x4*)(row + kbase + lh * 8);
  const f32x4* p1 = (const f32x4*)(row + kbase + 16 + lh * 8);
  f32x4 a0 = p0[0], a1 = p0[1], b0 = p1[0], b1 = p1[1];
  v16bf r;
#pragma unroll
  for (int i = 0; i < 4; ++i) {
    r[i]      = (bf16)a0[i];
    r[4 + i]  = (bf16)a1[i];
    r[8 + i]  = (bf16)b0[i];
    r[12 + i] = (bf16)b1[i];
  }
  return r;
}

// B-fragment (32x16, 16-bit) from LDS laid out as Bs[n][k], row stride `st`
// (bf16 elements, multiple of 8). Lane: n=L%16, element j maps to
// k = (j<8 ? lh*8 + j : 16 + lh*8 + (j-8)).
static __device__ __forceinline__ v16bf bfrag_lds(const bf16* base, int n, int lh, int st) {
  const bf16* p = base + n * st + lh * 8;
  bf16x8 lo = *(const bf16x8*)p;
  bf16x8 hi = *(const bf16x8*)(p + 16);
  v16bf r;
#pragma unroll
  for (int i = 0; i < 8; ++i) { r[i] = lo[i]; r[8 + i] = hi[i]; }
  return r;
}

// 16-lane (half-wave) reductions; wave32, xor masks < 16 stay in-half.
static __device__ __forceinline__ float redmax16(float v) {
#pragma unroll
  for (int m = 1; m < 16; m <<= 1) v = fmaxf(v, __shfl_xor(v, m, 32));
  return v;
}
static __device__ __forceinline__ float redsum16(float v) {
#pragma unroll
  for (int m = 1; m < 16; m <<= 1) v += __shfl_xor(v, m, 32);
  return v;
}

// ---------------------------------------------------------------------------
// Kernel 0: one-time fp32 -> bf16 conversion of Wf so outproj can stage its
// B tiles with async byte copies.
// ---------------------------------------------------------------------------
__global__ __launch_bounds__(256) void cvtWf_kernel(const float* __restrict__ W,
                                                    bf16* __restrict__ Wb) {
  size_t i = ((size_t)blockIdx.x * 256 + threadIdx.x) * 4;
  f32x4 x = *(const f32x4*)(W + i);
  bf16x4 y;
#pragma unroll
  for (int j = 0; j < 4; ++j) y[j] = (bf16)x[j];
  *(bf16x4*)(Wb + i) = y;
}

// ---------------------------------------------------------------------------
// Kernel 1: per-head QKV projection. out[b,h,s,e] = sum_d x[b,s,h,d]*W[e,d]
// ---------------------------------------------------------------------------
#define WSTRIDE 72
__global__ __launch_bounds__(256) void proj_kernel(
    const float* __restrict__ Xq, const float* __restrict__ Xk,
    const float* __restrict__ Xv, const float* __restrict__ Wq,
    const float* __restrict__ Wk, const float* __restrict__ Wv,
    bf16* __restrict__ Qp, bf16* __restrict__ Kp, bf16* __restrict__ Vp) {
  __shared__ bf16 Ws[64 * WSTRIDE];

  int bid  = blockIdx.x;
  int mat  = bid % 3;
  int rest = bid / 3;
  int qblk = rest & 15;
  int head = (rest >> 4) & 15;
  int b    = rest >> 8;

  const float* X   = (mat == 0) ? Xq : (mat == 1) ? Xk : Xv;
  const float* W   = (mat == 0) ? Wq : (mat == 1) ? Wk : Wv;
  bf16*        Out = (mat == 0) ? Qp : (mat == 1) ? Kp : Vp;

  // Stage W into LDS: Ws[e][d] = W[e*64+d], fp32 -> bf16 (needs cvt: sync).
  {
    int t  = threadIdx.x;
    int e  = t >> 2;
    int d0 = (t & 3) * 16;
    const f32x4* src = (const f32x4*)(W + e * 64 + d0);
#pragma unroll
    for (int c = 0; c < 2; ++c) {
      f32x4 x0 = src[2 * c], x1 = src[2 * c + 1];
      bf16x8 y;
#pragma unroll
      for (int i = 0; i < 4; ++i) { y[i] = (bf16)x0[i]; y[4 + i] = (bf16)x1[i]; }
      *(bf16x8*)&Ws[e * WSTRIDE + d0 + 8 * c] = y;
    }
  }
  __syncthreads();

  int lane = threadIdx.x & 31;
  int wave = threadIdx.x >> 5;
  int lh   = lane >> 4;
  int n    = lane & 15;

  int s = qblk * 128 + wave * 16 + n;  // A-fragment row (m = lane%16)
  const float* xrow = X + ((size_t)b * SS + s) * EE + head * DD;
  v16bf a0 = afrag_f32(xrow, 0, lh);
  v16bf a1 = afrag_f32(xrow, 32, lh);

  bf16* obase =
      Out + (((size_t)b * HH + head) * SS + qblk * 128 + wave * 16) * DD;
#pragma unroll
  for (int nt = 0; nt < 4; ++nt) {
    v8f acc = {};
    acc = wmma_bf16(a0, bfrag_lds(Ws + nt * 16 * WSTRIDE, n, lh, WSTRIDE), acc);
    acc = wmma_bf16(a1, bfrag_lds(Ws + nt * 16 * WSTRIDE + 32, n, lh, WSTRIDE), acc);
#pragma unroll
    for (int r = 0; r < 8; ++r) {
      int m = r + 8 * lh;
      obase[(size_t)m * DD + nt * 16 + n] = (bf16)acc[r];
    }
  }
}

// ---------------------------------------------------------------------------
// Kernel 2: flash attention over one (b, head, 128-query strip) per block.
// K tile staged via async-to-LDS; V staged through VGPRs (transpose needed).
// ---------------------------------------------------------------------------
#define KSTRIDE 72
#define VSTRIDE 40
__global__ __launch_bounds__(256) void attn_kernel(
    const bf16* __restrict__ Qp, const bf16* __restrict__ Kp,
    const bf16* __restrict__ Vp, const int* __restrict__ mask,
    bf16* __restrict__ ctx) {
  __shared__ bf16 Kt[32 * KSTRIDE];     // Kt[key][d]
  __shared__ bf16 Vt[64 * VSTRIDE];     // Vt[d][key] (transposed)
  __shared__ bf16 Pb[8 * 16 * 32];      // per-wave P staging (A-layout source)

  int bid  = blockIdx.x;
  int qblk = bid & 15;
  int head = (bid >> 4) & 15;
  int b    = bid >> 8;

  int lane = threadIdx.x & 31;
  int wave = threadIdx.x >> 5;
  int lh   = lane >> 4;
  int n    = lane & 15;

  size_t bh = (size_t)b * HH + head;
  int qrow_i = qblk * 128 + wave * 16 + n;
  const bf16* qrow = Qp + (bh * SS + qrow_i) * DD;
  v16bf qf0 = afrag_bf16(qrow, 0, lh);
  v16bf qf1 = afrag_bf16(qrow, 32, lh);

  const int* maskp = mask + (size_t)b * SS;

  v8f O0 = {}, O1 = {}, O2 = {}, O3 = {};
  float mrow[8], lrow[8];
#pragma unroll
  for (int r = 0; r < 8; ++r) { mrow[r] = NEG_INF; lrow[r] = 0.0f; }

  bf16* Pst  = Pb + wave * 512;        // store base (C-layout writes)
  bf16* Prow = Pst + n * 32;           // A-layout row for this lane (m=n)

  // Per-thread staging coordinates.
  int skey = threadIdx.x >> 3;
  int sd0  = (threadIdx.x & 7) * 8;

  for (int j = 0; j < SS; j += 32) {
    __syncthreads();
    {
      const bf16* ksrc = Kp + (bh * SS + j + skey) * DD + sd0;
#ifdef HAVE_ASYNC_LDS
      async_copy_b128(&Kt[skey * KSTRIDE + sd0], ksrc);
#else
      *(bf16x8*)&Kt[skey * KSTRIDE + sd0] = *(const bf16x8*)ksrc;
#endif
      bf16x8 vv = *(const bf16x8*)(Vp + (bh * SS + j + skey) * DD + sd0);
#pragma unroll
      for (int i = 0; i < 8; ++i) Vt[(sd0 + i) * VSTRIDE + skey] = vv[i];
    }
#ifdef HAVE_ASYNC_LDS
    wait_async0();
#endif
    __syncthreads();

    // S = Q * K^T  (two 16x16 tiles covering 32 keys)
    v8f S0 = {}, S1 = {};
    S0 = wmma_bf16(qf0, bfrag_lds(Kt, n, lh, KSTRIDE), S0);
    S0 = wmma_bf16(qf1, bfrag_lds(Kt + 32, n, lh, KSTRIDE), S0);
    S1 = wmma_bf16(qf0, bfrag_lds(Kt + 16 * KSTRIDE, n, lh, KSTRIDE), S1);
    S1 = wmma_bf16(qf1, bfrag_lds(Kt + 16 * KSTRIDE + 32, n, lh, KSTRIDE), S1);

    int mv0 = maskp[j + n];
    int mv1 = maskp[j + 16 + n];
#pragma unroll
    for (int r = 0; r < 8; ++r) {
      S0[r] = mv0 ? S0[r] * 0.125f : NEG_INF;
      S1[r] = mv1 ? S1[r] * 0.125f : NEG_INF;
    }

    // Online softmax per tracked row (VGPR r <-> row r + 8*lh).
#pragma unroll
    for (int r = 0; r < 8; ++r) {
      float mx   = redmax16(fmaxf(S0[r], S1[r]));
      float aold = mrow[r];
      float mnew = fmaxf(aold, mx);
      mrow[r]    = mnew;
      float alpha = (aold == NEG_INF) ? 0.0f : __expf(aold - mnew);
      float p0 = (mnew == NEG_INF) ? 0.0f : __expf(S0[r] - mnew);
      float p1 = (mnew == NEG_INF) ? 0.0f : __expf(S1[r] - mnew);
      lrow[r] = alpha * lrow[r] + redsum16(p0 + p1);
      O0[r] *= alpha; O1[r] *= alpha; O2[r] *= alpha; O3[r] *= alpha;
      int m = r + 8 * lh;
      Pst[m * 32 + n]      = (bf16)p0;   // C-layout -> A-layout via LDS
      Pst[m * 32 + 16 + n] = (bf16)p1;
    }

    // Same-wave LDS RAW: DS ops are in-order within a wave.
    v16bf pf = afrag_bf16(Prow, 0, lh);
    O0 = wmma_bf16(pf, bfrag_lds(Vt, n, lh, VSTRIDE), O0);
    O1 = wmma_bf16(pf, bfrag_lds(Vt + 16 * VSTRIDE, n, lh, VSTRIDE), O1);
    O2 = wmma_bf16(pf, bfrag_lds(Vt + 32 * VSTRIDE, n, lh, VSTRIDE), O2);
    O3 = wmma_bf16(pf, bfrag_lds(Vt + 48 * VSTRIDE, n, lh, VSTRIDE), O3);

    if (j + 32 < SS) {  // global_prefetch_b8 for next tiles
      __builtin_prefetch(Kp + (bh * SS + j + 32) * DD, 0, 0);
      __builtin_prefetch(Vp + (bh * SS + j + 32) * DD, 0, 0);
    }
  }

  // Epilogue: normalize and write ctx[b, q, head*64 + d] as bf16.
  bf16* obase =
      ctx + ((size_t)b * SS + qblk * 128 + wave * 16) * EE + head * DD;
#pragma unroll
  for (int r = 0; r < 8; ++r) {
    float inv = (lrow[r] > 0.0f) ? 1.0f / lrow[r] : 0.0f;
    int m = r + 8 * lh;
    bf16* orow = obase + (size_t)m * EE;
    orow[n]      = (bf16)(O0[r] * inv);
    orow[16 + n] = (bf16)(O1[r] * inv);
    orow[32 + n] = (bf16)(O2[r] * inv);
    orow[48 + n] = (bf16)(O3[r] * inv);
  }
}

// ---------------------------------------------------------------------------
// Kernel 3: output projection out = ctx @ Wf^T + bf (8192x1024 @ 1024x1024).
// B tile (Wfb, bf16) staged with double-buffered async-to-LDS copies.
// ---------------------------------------------------------------------------
#define BSTRIDE 40
__global__ __launch_bounds__(256) void outproj_kernel(
    const bf16* __restrict__ ctx, const bf16* __restrict__ Wfb,
    const float* __restrict__ bias, float* __restrict__ out) {
  __shared__ bf16 Bs[2][64 * BSTRIDE];  // Bs[buf][e'][f'] = Wfb[nbase+e', kb+f']

  int bid  = blockIdx.x;
  int nblk = bid & 15;
  int mblk = bid >> 4;

  int lane = threadIdx.x & 31;
  int wave = threadIdx.x >> 5;
  int lh   = lane >> 4;
  int n    = lane & 15;

  int nbase = nblk * 64;
  int arow  = mblk * 128 + wave * 16 + n;  // A-fragment row (m = lane%16)
  const bf16* actx = ctx + (size_t)arow * EE;

  // Per-thread staging coords: one b128 (8 bf16) per tile per thread.
  int se = threadIdx.x >> 2;
  int sf = (threadIdx.x & 3) * 8;
  const bf16* wsrc = Wfb + (size_t)(nbase + se) * EE + sf;
  bf16* sdst0 = &Bs[0][se * BSTRIDE + sf];
  bf16* sdst1 = &Bs[1][se * BSTRIDE + sf];

  v8f acc0 = {}, acc1 = {}, acc2 = {}, acc3 = {};

#ifdef HAVE_ASYNC_LDS
  // Double-buffered async pipeline: async loads complete in-order per wave,
  // so s_wait_asynccnt<=1 guarantees the current tile has landed.
  async_copy_b128(sdst0, wsrc);
  for (int kb = 0, i = 0; kb < EE; kb += 32, ++i) {
    int cur = i & 1;
    bool more = (kb + 32) < EE;
    if (more) async_copy_b128(cur ? sdst0 : sdst1, wsrc + kb + 32);
    if (more) wait_async1(); else wait_async0();
    __syncthreads();   // all waves' tile-i copies visible
    const bf16* Bt = &Bs[cur][0];
    v16bf a = afrag_bf16(actx, kb, lh);
    acc0 = wmma_bf16(a, bfrag_lds(Bt, n, lh, BSTRIDE), acc0);
    acc1 = wmma_bf16(a, bfrag_lds(Bt + 16 * BSTRIDE, n, lh, BSTRIDE), acc1);
    acc2 = wmma_bf16(a, bfrag_lds(Bt + 32 * BSTRIDE, n, lh, BSTRIDE), acc2);
    acc3 = wmma_bf16(a, bfrag_lds(Bt + 48 * BSTRIDE, n, lh, BSTRIDE), acc3);
    __syncthreads();   // everyone done reading before buffer reuse
  }
#else
  for (int kb = 0; kb < EE; kb += 32) {
    __syncthreads();
    *(bf16x8*)sdst0 = *(const bf16x8*)(wsrc + kb);
    __syncthreads();
    const bf16* Bt = &Bs[0][0];
    v16bf a = afrag_bf16(actx, kb, lh);
    acc0 = wmma_bf16(a, bfrag_lds(Bt, n, lh, BSTRIDE), acc0);
    acc1 = wmma_bf16(a, bfrag_lds(Bt + 16 * BSTRIDE, n, lh, BSTRIDE), acc1);
    acc2 = wmma_bf16(a, bfrag_lds(Bt + 32 * BSTRIDE, n, lh, BSTRIDE), acc2);
    acc3 = wmma_bf16(a, bfrag_lds(Bt + 48 * BSTRIDE, n, lh, BSTRIDE), acc3);
  }
#endif

  float b0 = bias[nbase + n];
  float b1 = bias[nbase + 16 + n];
  float b2 = bias[nbase + 32 + n];
  float b3 = bias[nbase + 48 + n];
  int orow0 = mblk * 128 + wave * 16;
#pragma unroll
  for (int r = 0; r < 8; ++r) {
    int m = r + 8 * lh;
    float* orow = out + (size_t)(orow0 + m) * EE + nbase;
    orow[n]      = acc0[r] + b0;
    orow[16 + n] = acc1[r] + b1;
    orow[32 + n] = acc2[r] + b2;
    orow[48 + n] = acc3[r] + b3;
  }
}

// ---------------------------------------------------------------------------
extern "C" void kernel_launch(void* const* d_in, const int* in_sizes, int n_in,
                              void* d_out, int out_size, void* d_ws,
                              size_t ws_size, hipStream_t stream) {
  const float* query = (const float*)d_in[0];
  const float* keys  = (const float*)d_in[1];
  const float* vals  = (const float*)d_in[2];
  const int*   mask  = (const int*)d_in[3];
  const float* Wq    = (const float*)d_in[4];
  const float* Wk    = (const float*)d_in[5];
  const float* Wv    = (const float*)d_in[6];
  const float* Wf    = (const float*)d_in[7];
  const float* bf    = (const float*)d_in[8];
  float*       out   = (float*)d_out;

  // Workspace: Qp | Kp | Vp | ctx (each 8Mi bf16 = 16 MiB) | Wfb (2 MiB).
  size_t PER = (size_t)BB * HH * SS * DD;
  bf16* ws  = (bf16*)d_ws;
  bf16* Qp  = ws;
  bf16* Kp  = ws + PER;
  bf16* Vp  = ws + 2 * PER;
  bf16* ctx = ws + 3 * PER;
  bf16* Wfb = ws + 4 * PER;

  cvtWf_kernel<<<(EE * EE) / (256 * 4), 256, 0, stream>>>(Wf, Wfb);
  proj_kernel<<<3 * BB * HH * (SS / 128), 256, 0, stream>>>(
      query, keys, vals, Wq, Wk, Wv, Qp, Kp, Vp);
  attn_kernel<<<BB * HH * (SS / 128), 256, 0, stream>>>(Qp, Kp, Vp, mask, ctx);
  outproj_kernel<<<(BB * SS / 128) * (EE / 64), 256, 0, stream>>>(
      ctx, Wfb, bf, out);
}
